// Wavenet_10428180595000
// MI455X (gfx1250) — matrix operand order repaired and checked by reading, not compiled, not run on hardware
//
#include <hip/hip_runtime.h>

typedef __bf16 bf16_t;
typedef __attribute__((ext_vector_type(16))) __bf16 v16bf;
typedef __attribute__((ext_vector_type(8)))  float  v8f;
typedef __attribute__((ext_vector_type(4)))  int    v4i;

#define TN 64          // time-steps per block tile
#define RS 136         // LDS row stride (bf16 elems), padded vs 128 to spread banks
#define CH 128         // residual channels
#define TT 16384       // sequence length
#define QD 256         // output channels

union Frag16 {
  v16bf v;
  struct { v4i lo; v4i hi; } q;
};

// A fragment (16x32 MxK, bf16): lane holds row M=lane%16.
// lanes 0-15: K in {k0..k0+7, k0+16..k0+23}; lanes 16-31: +8.
static __device__ __forceinline__ v16bf load_frag_a(const bf16_t* p) {
  Frag16 f;
  f.q.lo = *(const v4i*)(p);        // 8 bf16 = 16B
  f.q.hi = *(const v4i*)(p + 16);   // next K half, +32B
  return f.v;
}
// B fragment (32x16 KxN, bf16): lane holds col N=lane%16, K in {k0..k0+7, k0+16..k0+23} (+8 for lanes 16-31).
static __device__ __forceinline__ v16bf load_frag_b(const bf16_t* p) {
  Frag16 f;
  f.q.lo = *(const v4i*)(p);
  f.q.hi = *(const v4i*)(p + 16);
  return f.v;
}
static __device__ __forceinline__ v8f v8f_zero() {
  v8f z = {0.f,0.f,0.f,0.f,0.f,0.f,0.f,0.f};
  return z;
}
static __device__ __forceinline__ v8f wmma_bf16(v16bf a, v16bf b, v8f c) {
  return __builtin_amdgcn_wmma_f32_16x16x32_bf16(false, a, false, b, (short)0,
                                                 c, false, false);
}

// ---------------- weight prep ----------------

__global__ void cvt_bf16_kernel(const float* __restrict__ src,
                                bf16_t* __restrict__ dst, size_t n) {
  size_t i = (size_t)blockIdx.x * blockDim.x + threadIdx.x;
  const size_t stride = (size_t)gridDim.x * blockDim.x;
  for (; i < n; i += stride) dst[i] = (bf16_t)src[i];
}

// dcnn_w [L][O][I][3] fp32  ->  [L][tap][O][I] bf16
__global__ void reorder_dcnn_kernel(const float* __restrict__ src,
                                    bf16_t* __restrict__ dst) {
  const size_t n = (size_t)63 * 3 * CH * CH;
  size_t idx = (size_t)blockIdx.x * blockDim.x + threadIdx.x;
  const size_t stride = (size_t)gridDim.x * blockDim.x;
  for (; idx < n; idx += stride) {
    const int ic = (int)(idx % CH);
    size_t r = idx / CH;
    const int o = (int)(r % CH); r /= CH;
    const int k = (int)(r % 3);
    const size_t l = r / 3;
    dst[idx] = (bf16_t)src[(((l * CH + o) * CH) + ic) * 3 + k];
  }
}

__global__ void zero_kernel(float* __restrict__ p, size_t n) {
  size_t i = (size_t)blockIdx.x * blockDim.x + threadIdx.x;
  const size_t stride = (size_t)gridDim.x * blockDim.x;
  for (; i < n; i += stride) p[i] = 0.0f;
}

// ---------------- front causal conv (1 -> 128 ch, k=25, pad 12) ----------------

__global__ __launch_bounds__(256) void causal_conv_kernel(
    const float* __restrict__ x, const float* __restrict__ cw,
    const float* __restrict__ cb, float* __restrict__ xout) {
  __shared__ float wsh[25];
  __shared__ float xsh[256 + 24];
  const int tid = threadIdx.x;
  const int c = blockIdx.y;
  const int b = blockIdx.z;
  const int t0 = blockIdx.x * 256;
  if (tid < 25) wsh[tid] = cw[c * 25 + tid];
  const float* xb = x + (size_t)b * TT;
  for (int i = tid; i < 256 + 24; i += 256) {
    const int tg = t0 + i - 12;
    xsh[i] = (tg >= 0 && tg < TT) ? xb[tg] : 0.0f;
  }
  __syncthreads();
  float acc = cb[c];
#pragma unroll
  for (int j = 0; j < 25; ++j) acc += wsh[j] * xsh[tid + j];
  xout[((size_t)b * CH + c) * TT + t0 + tid] = acc;
}

// ---------------- one WaveNet layer (fused dconv + dense + skip) ----------------
// Wave register block: 2 M-tiles x 2 N-tiles (32x32 output), so B fragments are
// reused across 2 WMMAs -> 1 ds_load_b128 per WMMA (LDS-bandwidth headroom).

__global__ __launch_bounds__(256) void wavenet_layer_kernel(
    const float* __restrict__ xin, float* __restrict__ xout,
    float* __restrict__ skip,
    const bf16_t* __restrict__ wdc,   // [3][128][128] bf16
    const bf16_t* __restrict__ wdn,   // [128][128] bf16
    const bf16_t* __restrict__ wsk,   // [128][128] bf16 (aliased when unused)
    const float* __restrict__ bdc, const float* __restrict__ bdn,
    const float* __restrict__ bsk,
    int d, int has_skip) {
  __shared__ bf16_t xs[3][TN * RS];   // relu(x) tiles at t-d, t, t+d (transposed [t][c])
  __shared__ bf16_t hs[TN * RS];      // relu(dconv) tile (transposed [t][c])

  const int tid = threadIdx.x;
  const int b = blockIdx.y;
  const int t0 = blockIdx.x * TN;
  const size_t bofs = (size_t)b * CH * TT;
  const float* xb = xin + bofs;

  // ---- stage 1: fill three shifted relu(x) tiles ----
  for (int tap = 0; tap < 3; ++tap) {
    const int dt = (tap - 1) * d;
    for (int i = tid; i < CH * TN; i += 256) {
      const int c  = i >> 6;
      const int tt = i & 63;
      const int tg = t0 + tt + dt;
      float v = 0.0f;
      if (tg >= 0 && tg < TT) v = xb[(size_t)c * TT + tg];
      v = fmaxf(v, 0.0f);
      xs[tap][tt * RS + c] = (bf16_t)v;
    }
  }
  __syncthreads();

  const int wv = tid >> 5;
  const int l  = tid & 31;
  const int lr = l & 15;
  const int hi = l >> 4;         // lane half-group
  const int mg = wv >> 1;        // M-group: rows [mg*32, mg*32+32)
  const int ng = wv & 1;         // N-group: cols [ng*32, ng*32+32)
  const int Mb = mg * 32;
  const int Nb = ng * 32;

  // ---- stage 2: dilated conv GEMM, M=128 K=3*128 N=64 ----
  v8f acc[2][2];
#pragma unroll
  for (int mi = 0; mi < 2; ++mi)
#pragma unroll
    for (int ni = 0; ni < 2; ++ni) acc[mi][ni] = v8f_zero();

#pragma unroll
  for (int tap = 0; tap < 3; ++tap) {
    const bf16_t* wt = wdc + (size_t)tap * CH * CH;
#pragma unroll
    for (int kc = 0; kc < 4; ++kc) {
      const int k0 = kc * 32;
      const v16bf a0 = load_frag_a(wt + (size_t)(Mb + lr) * CH + k0 + hi * 8);
      const v16bf a1 = load_frag_a(wt + (size_t)(Mb + 16 + lr) * CH + k0 + hi * 8);
      const v16bf b0 = load_frag_b(&xs[tap][(Nb + lr) * RS + k0 + hi * 8]);
      const v16bf b1 = load_frag_b(&xs[tap][(Nb + 16 + lr) * RS + k0 + hi * 8]);
      acc[0][0] = wmma_bf16(a0, b0, acc[0][0]);
      acc[0][1] = wmma_bf16(a0, b1, acc[0][1]);
      acc[1][0] = wmma_bf16(a1, b0, acc[1][0]);
      acc[1][1] = wmma_bf16(a1, b1, acc[1][1]);
    }
  }

  // bias + relu -> hs
#pragma unroll
  for (int mi = 0; mi < 2; ++mi) {
#pragma unroll
    for (int ni = 0; ni < 2; ++ni) {
      const int t = Nb + ni * 16 + lr;
#pragma unroll
      for (int r = 0; r < 8; ++r) {
        const int c = Mb + mi * 16 + r + hi * 8;
        float v = acc[mi][ni][r] + bdc[c];
        v = fmaxf(v, 0.0f);
        hs[t * RS + c] = (bf16_t)v;
      }
    }
  }
  __syncthreads();

  // ---- stage 3: dense (+skip) GEMMs, shared B fragments ----
  v8f aD[2][2], aS[2][2];
#pragma unroll
  for (int mi = 0; mi < 2; ++mi)
#pragma unroll
    for (int ni = 0; ni < 2; ++ni) { aD[mi][ni] = v8f_zero(); aS[mi][ni] = v8f_zero(); }

#pragma unroll
  for (int kc = 0; kc < 4; ++kc) {
    const int k0 = kc * 32;
    const v16bf ad0 = load_frag_a(wdn + (size_t)(Mb + lr) * CH + k0 + hi * 8);
    const v16bf ad1 = load_frag_a(wdn + (size_t)(Mb + 16 + lr) * CH + k0 + hi * 8);
    v16bf as0 = ad0, as1 = ad1;
    if (has_skip) {
      as0 = load_frag_a(wsk + (size_t)(Mb + lr) * CH + k0 + hi * 8);
      as1 = load_frag_a(wsk + (size_t)(Mb + 16 + lr) * CH + k0 + hi * 8);
    }
    const v16bf b0 = load_frag_b(&hs[(Nb + lr) * RS + k0 + hi * 8]);
    const v16bf b1 = load_frag_b(&hs[(Nb + 16 + lr) * RS + k0 + hi * 8]);
    aD[0][0] = wmma_bf16(ad0, b0, aD[0][0]);
    aD[0][1] = wmma_bf16(ad0, b1, aD[0][1]);
    aD[1][0] = wmma_bf16(ad1, b0, aD[1][0]);
    aD[1][1] = wmma_bf16(ad1, b1, aD[1][1]);
    if (has_skip) {
      aS[0][0] = wmma_bf16(as0, b0, aS[0][0]);
      aS[0][1] = wmma_bf16(as0, b1, aS[0][1]);
      aS[1][0] = wmma_bf16(as1, b0, aS[1][0]);
      aS[1][1] = wmma_bf16(as1, b1, aS[1][1]);
    }
  }

  float* xob = xout + bofs;
  float* skb = skip + bofs;
#pragma unroll
  for (int mi = 0; mi < 2; ++mi) {
#pragma unroll
    for (int ni = 0; ni < 2; ++ni) {
      const int tg = t0 + Nb + ni * 16 + lr;
#pragma unroll
      for (int r = 0; r < 8; ++r) {
        const int c = Mb + mi * 16 + r + hi * 8;
        const size_t idx = (size_t)c * TT + tg;
        xob[idx] = aD[mi][ni][r] + bdn[c] + xb[idx];       // residual (fp32)
        if (has_skip) skb[idx] += aS[mi][ni][r] + bsk[c];  // skip accumulate
      }
    }
  }
}

// ---------------- post network: relu -> 128x128 -> relu -> 256x128 ----------------

__global__ __launch_bounds__(256) void wavenet_post_kernel(
    const float* __restrict__ skip,
    const bf16_t* __restrict__ wp1, const float* __restrict__ bp1,
    const bf16_t* __restrict__ wp2, const float* __restrict__ bp2,
    float* __restrict__ out) {
  __shared__ bf16_t ss[TN * RS];
  __shared__ bf16_t hs[TN * RS];
  const int tid = threadIdx.x;
  const int b = blockIdx.y;
  const int t0 = blockIdx.x * TN;
  const float* skb = skip + (size_t)b * CH * TT;

  for (int i = tid; i < CH * TN; i += 256) {
    const int c  = i >> 6;
    const int tt = i & 63;
    const float v = skb[(size_t)c * TT + t0 + tt];
    ss[tt * RS + c] = (bf16_t)fmaxf(v, 0.0f);
  }
  __syncthreads();

  const int wv = tid >> 5;
  const int l  = tid & 31;
  const int lr = l & 15;
  const int hi = l >> 4;
  const int mg = wv >> 1;
  const int ng = wv & 1;
  const int Mb = mg * 32;
  const int Nb = ng * 32;

  // GEMM1: h = relu(W1 * relu(skip) + b1), M=128 K=128
  v8f acc[2][2];
#pragma unroll
  for (int mi = 0; mi < 2; ++mi)
#pragma unroll
    for (int ni = 0; ni < 2; ++ni) acc[mi][ni] = v8f_zero();
#pragma unroll
  for (int kc = 0; kc < 4; ++kc) {
    const int k0 = kc * 32;
    const v16bf a0 = load_frag_a(wp1 + (size_t)(Mb + lr) * CH + k0 + hi * 8);
    const v16bf a1 = load_frag_a(wp1 + (size_t)(Mb + 16 + lr) * CH + k0 + hi * 8);
    const v16bf b0 = load_frag_b(&ss[(Nb + lr) * RS + k0 + hi * 8]);
    const v16bf b1 = load_frag_b(&ss[(Nb + 16 + lr) * RS + k0 + hi * 8]);
    acc[0][0] = wmma_bf16(a0, b0, acc[0][0]);
    acc[0][1] = wmma_bf16(a0, b1, acc[0][1]);
    acc[1][0] = wmma_bf16(a1, b0, acc[1][0]);
    acc[1][1] = wmma_bf16(a1, b1, acc[1][1]);
  }
#pragma unroll
  for (int mi = 0; mi < 2; ++mi) {
#pragma unroll
    for (int ni = 0; ni < 2; ++ni) {
      const int t = Nb + ni * 16 + lr;
#pragma unroll
      for (int r = 0; r < 8; ++r) {
        const int c = Mb + mi * 16 + r + hi * 8;
        float v = acc[mi][ni][r] + bp1[c];
        v = fmaxf(v, 0.0f);
        hs[t * RS + c] = (bf16_t)v;
      }
    }
  }
  __syncthreads();

  // GEMM2: out = W2 * h + b2, M=256 K=128 (each wave does 2 M-groups of 32)
  float* ob = out + (size_t)b * QD * TT;
  for (int mo = 0; mo < 2; ++mo) {
    const int Qb = Mb + mo * 128;
    v8f a2[2][2];
#pragma unroll
    for (int mi = 0; mi < 2; ++mi)
#pragma unroll
      for (int ni = 0; ni < 2; ++ni) a2[mi][ni] = v8f_zero();
#pragma unroll
    for (int kc = 0; kc < 4; ++kc) {
      const int k0 = kc * 32;
      const v16bf a0 = load_frag_a(wp2 + (size_t)(Qb + lr) * CH + k0 + hi * 8);
      const v16bf a1 = load_frag_a(wp2 + (size_t)(Qb + 16 + lr) * CH + k0 + hi * 8);
      const v16bf b0 = load_frag_b(&hs[(Nb + lr) * RS + k0 + hi * 8]);
      const v16bf b1 = load_frag_b(&hs[(Nb + 16 + lr) * RS + k0 + hi * 8]);
      a2[0][0] = wmma_bf16(a0, b0, a2[0][0]);
      a2[0][1] = wmma_bf16(a0, b1, a2[0][1]);
      a2[1][0] = wmma_bf16(a1, b0, a2[1][0]);
      a2[1][1] = wmma_bf16(a1, b1, a2[1][1]);
    }
#pragma unroll
    for (int mi = 0; mi < 2; ++mi) {
#pragma unroll
      for (int ni = 0; ni < 2; ++ni) {
        const int tg = t0 + Nb + ni * 16 + lr;
#pragma unroll
        for (int r = 0; r < 8; ++r) {
          const int q = Qb + mi * 16 + r + hi * 8;
          ob[(size_t)q * TT + tg] = a2[mi][ni][r] + bp2[q];
        }
      }
    }
  }
}

// ---------------- host launch ----------------

extern "C" void kernel_launch(void* const* d_in, const int* in_sizes, int n_in,
                              void* d_out, int out_size, void* d_ws, size_t ws_size,
                              hipStream_t stream) {
  (void)in_sizes; (void)n_in; (void)out_size; (void)ws_size;

  const float* x        = (const float*)d_in[0];
  const float* causal_w = (const float*)d_in[1];
  const float* causal_b = (const float*)d_in[2];
  const float* dcnn_w   = (const float*)d_in[3];
  const float* dcnn_b   = (const float*)d_in[4];
  const float* dense_w  = (const float*)d_in[5];
  const float* dense_b  = (const float*)d_in[6];
  const float* skip_w   = (const float*)d_in[7];
  const float* skip_b   = (const float*)d_in[8];
  const float* post1_w  = (const float*)d_in[9];
  const float* post1_b  = (const float*)d_in[10];
  const float* post2_w  = (const float*)d_in[11];
  const float* post2_b  = (const float*)d_in[12];

  const int Bn = 8, L = 63, N0 = 27;
  const size_t NX = (size_t)Bn * CH * TT;

  char* ws = (char*)d_ws;
  size_t off = 0;
  auto take = [&](size_t bytes) -> char* {
    char* p = ws + off;
    off += (bytes + 255) & ~(size_t)255;
    return p;
  };
  float*  xA   = (float*)take(NX * 4);
  float*  xB   = (float*)take(NX * 4);
  float*  skip = (float*)take(NX * 4);
  bf16_t* wdc  = (bf16_t*)take((size_t)L * 3 * CH * CH * 2);
  bf16_t* wdn  = (bf16_t*)take((size_t)L * CH * CH * 2);
  bf16_t* wsk  = (bf16_t*)take((size_t)L * CH * CH * 2);
  bf16_t* wp1  = (bf16_t*)take((size_t)CH * CH * 2);
  bf16_t* wp2  = (bf16_t*)take((size_t)QD * CH * 2);

  // weight prep (bf16 conversion / dcnn reorder)
  reorder_dcnn_kernel<<<2048, 256, 0, stream>>>(dcnn_w, wdc);
  cvt_bf16_kernel<<<1024, 256, 0, stream>>>(dense_w, wdn, (size_t)L * CH * CH);
  cvt_bf16_kernel<<<1024, 256, 0, stream>>>(skip_w,  wsk, (size_t)L * CH * CH);
  cvt_bf16_kernel<<<64,   256, 0, stream>>>(post1_w, wp1, (size_t)CH * CH);
  cvt_bf16_kernel<<<64,   256, 0, stream>>>(post2_w, wp2, (size_t)QD * CH);

  zero_kernel<<<2048, 256, 0, stream>>>(skip, NX);

  dim3 cgrid(TT / 256, CH, Bn);
  causal_conv_kernel<<<cgrid, 256, 0, stream>>>(x, causal_w, causal_b, xA);

  dim3 lgrid(TT / TN, Bn);
  float* cur = xA;
  float* nxt = xB;
  for (int li = 0; li < L; ++li) {
    const int d = 1 << (li % 9);
    const int has_skip = (li >= N0) ? 1 : 0;
    const bf16_t* wdnl = wdn + (size_t)li * CH * CH;
    const bf16_t* wskl = has_skip ? (wsk + (size_t)li * CH * CH) : wdnl;
    wavenet_layer_kernel<<<lgrid, 256, 0, stream>>>(
        cur, nxt, skip,
        wdc + (size_t)li * 3 * CH * CH, wdnl, wskl,
        dcnn_b + (size_t)li * CH,
        dense_b + (size_t)li * CH,
        skip_b + (size_t)li * CH,
        d, has_skip);
    float* tmp = cur; cur = nxt; nxt = tmp;
  }

  wavenet_post_kernel<<<lgrid, 256, 0, stream>>>(
      skip, wp1, post1_b, wp2, post2_b, (float*)d_out);
}
